// PermInvariantQNN_618475290799
// MI455X (gfx1250) — compile-verified, matrix operand order checked by reading, not run on hardware
//
#include <hip/hip_runtime.h>

typedef _Float16 half_t;
typedef half_t v16h __attribute__((ext_vector_type(16)));
typedef half_t v8h  __attribute__((ext_vector_type(8)));
typedef float  v8f  __attribute__((ext_vector_type(8)));

#define NB   64   // number of invariant blocks (block size 1)
#define HID  20
#define NM   5
#define NIN  8
#define ODIM 4
#define WAVES 8

// Fragment K index for 16-bit A/B operands of v_wmma_*_16x16x32 (ISA 7.12.2):
// lanes 0-15 hold K = {0..7, 16..23}; lanes 16-31 hold K = {8..15, 24..31}
__device__ __forceinline__ int kmap(int s, int lh) {
  return ((s >> 3) << 4) + (lh << 3) + (s & 7);
}

__device__ __forceinline__ v8f splat8(float x) {
  v8f r;
#pragma unroll
  for (int i = 0; i < 8; ++i) r[i] = x;
  return r;
}

// Single-instruction relu: v_med3_f32(x, 0, +inf) == clamp(x, 0, inf)
__device__ __forceinline__ float relu1(float x) {
  return __builtin_amdgcn_fmed3f(x, 0.f, __builtin_inff());
}

__device__ __forceinline__ v8f relu8(v8f x) {
  v8f r;
#pragma unroll
  for (int i = 0; i < 8; ++i) r[i] = relu1(x[i]);
  return r;
}

// Build B fragment (K x 16 column tile starting at colBase), zero-padded to K=32
__device__ __forceinline__ v16h load_bfrag(const float* W, int K, int N, int colBase,
                                           int lane, float scale) {
  const int lcol = lane & 15, lh = lane >> 4;
  const int n = colBase + lcol;
  v16h f;
#pragma unroll
  for (int s = 0; s < 16; ++s) {
    int k = kmap(s, lh);
    float v = (k < K && n < N) ? (W[k * N + n] * scale) : 0.f;
    f[s] = (half_t)v;
  }
  return f;
}

// reuse_a must be an immediate -> template parameter
template <bool REUSE_A>
__device__ __forceinline__ v8f wmma16(v16h a, v16h b, v8f c) {
  // (neg_a, A, neg_b, B, c_mod, C, reuse_a, reuse_b)
  return __builtin_amdgcn_wmma_f32_16x16x32_f16(false, a, false, b, (short)0, c, REUSE_A, false);
}

// Write a 16x16(+16x16) f32 C tile pair into LDS as a row-major 16x32 f16 tile
__device__ __forceinline__ void store_ctile(half_t (*scw)[32], v8f t0, v8f t1, int lane, int nvalid) {
  const int lcol = lane & 15, lh = lane >> 4;
#pragma unroll
  for (int v = 0; v < 8; ++v) {
    int m = lh * 8 + v;                         // C layout: VGPR v -> row (lh*8 + v)
    scw[m][lcol]      = (half_t)t0[v];
    scw[m][16 + lcol] = (half_t)((16 + lcol < nvalid) ? t1[v] : 0.f);
  }
}

// Read a 16x32 row-major f16 LDS tile back as an A fragment (two b128 DS loads)
__device__ __forceinline__ v16h load_afrag(const half_t (*scw)[32], int lane) {
  const int lcol = lane & 15, lh = lane >> 4;
  const v8h* p = (const v8h*)&scw[lcol][0];
  v8h lo = p[lh];        // K = lh*8 .. lh*8+7
  v8h hi = p[2 + lh];    // K = 16 + lh*8 .. +7
  v16h a;
#pragma unroll
  for (int i = 0; i < 8; ++i) { a[i] = lo[i]; a[8 + i] = hi[i]; }
  return a;
}

__global__ __launch_bounds__(256, 1)
void perminv_qnn_kernel(const float* __restrict__ invar,
                        const float* __restrict__ ninv,
                        const float* __restrict__ eW1, const float* __restrict__ eb1,
                        const float* __restrict__ eW2, const float* __restrict__ eb2,
                        const float* __restrict__ eW3, const float* __restrict__ eb3,
                        const float* __restrict__ dW1, const float* __restrict__ db1,
                        const float* __restrict__ dW2, const float* __restrict__ db2,
                        const float* __restrict__ dW3, const float* __restrict__ db3,
                        float* __restrict__ out) {
  __shared__ half_t sx[WAVES][NB][16];                      // transposed f16 input tile per wave
  __shared__ __align__(64) half_t sc[WAVES][16][32];        // C->A transpose scratch per wave

  const int tid  = threadIdx.x;
  const int wave = tid >> 5;
  const int lane = tid & 31;
  const int lcol = lane & 15;
  const int lh   = lane >> 4;
  const int r0   = (blockIdx.x * WAVES + wave) * 16;        // first batch row of this wave

  // ---- stage 16x64 f32 input tile into LDS transposed + converted: sx[block][row] ----
  {
    const float4* rp = (const float4*)(invar + (size_t)(r0 + lcol) * NB + lh * 32);
#pragma unroll
    for (int q = 0; q < 8; ++q) {
      float4 d = rp[q];
      int c0 = lh * 32 + q * 4;
      sx[wave][c0 + 0][lcol] = (half_t)d.x;
      sx[wave][c0 + 1][lcol] = (half_t)d.y;
      sx[wave][c0 + 2][lcol] = (half_t)d.z;
      sx[wave][c0 + 3][lcol] = (half_t)d.w;
    }
  }
  __builtin_prefetch(ninv + (size_t)r0 * NIN, 0, 0);        // global_prefetch_b8

  // ---- per-lane layer-1 coefficients, pre-arranged in A-fragment order ----
  v16h w1h, b1h;
#pragma unroll
  for (int s = 0; s < 16; ++s) {
    int k = kmap(s, lh);
    w1h[s] = (half_t)((k < HID) ? eW1[k] : 0.f);
    b1h[s] = (half_t)((k < HID) ? eb1[k] : 0.f);
  }

  // ---- B fragments (weights), zero-padded; resident in VGPRs ----
  // 1/NB mean factor folded into the eW3 fragment: mean(h2)@W3 == sum(h2)@(W3/NB)
  const float inv_nb = 1.f / (float)NB;
  v16h bW2a = load_bfrag(eW2, HID, HID, 0, lane, 1.f);
  v16h bW2b = load_bfrag(eW2, HID, HID, 16, lane, 1.f);
  v16h bW3  = load_bfrag(eW3, HID, NM, 0, lane, inv_nb);
  v16h bD1a = load_bfrag(dW1, NM + NIN, HID, 0, lane, 1.f);
  v16h bD1b = load_bfrag(dW1, NM + NIN, HID, 16, lane, 1.f);
  v16h bD2a = load_bfrag(dW2, HID, HID, 0, lane, 1.f);
  v16h bD2b = load_bfrag(dW2, HID, HID, 16, lane, 1.f);
  v16h bD3  = load_bfrag(dW3, HID, ODIM, 0, lane, 1.f);

  // ---- bias tiles folded into the WMMA C operand ----
  v8f cb2a = splat8(eb2[lcol]);
  v8f cb2b = splat8((16 + lcol < HID) ? eb2[16 + lcol] : 0.f);
  v8f cb3  = splat8((lcol < NM) ? eb3[lcol] : 0.f);
  v8f cd1a = splat8(db1[lcol]);
  v8f cd1b = splat8((16 + lcol < HID) ? db1[16 + lcol] : 0.f);
  v8f cd2a = splat8(db2[lcol]);
  v8f cd2b = splat8((16 + lcol < HID) ? db2[16 + lcol] : 0.f);
  v8f cd3  = splat8((lcol < ODIM) ? db3[lcol] : 0.f);

  __syncthreads();

  // ---- encoder: accumulate sum(relu(h1 @ eW2 + eb2)) over 64 blocks ----
  v8f hs0 = splat8(0.f), hs1 = splat8(0.f);
#pragma unroll 4
  for (int b = 0; b < NB; ++b) {
    half_t xh = sx[wave][b][lcol];                          // ds_load_u16 broadcast
    v16h a;
#pragma unroll
    for (int s = 0; s < 16; ++s) {                          // h1 = relu(x*W1 + b1), f16 packed math
      half_t t = xh * w1h[s] + b1h[s];
      a[s] = (t > (half_t)0) ? t : (half_t)0;
    }
    v8f t0 = wmma16<false>(a, bW2a, cb2a);                  // 2x v_wmma_f32_16x16x32_f16
    v8f t1 = wmma16<true>(a, bW2b, cb2b);                   // reuse_a: same A as previous WMMA
#pragma unroll
    for (int i = 0; i < 8; ++i) {                           // v_med3_f32 relu + v_pk_add_f32
      hs0[i] += relu1(t0[i]);
      hs1[i] += relu1(t1[i]);
    }
  }

  // ---- moments = sum(h2) @ (eW3/NB) + eb3 ----
  store_ctile(sc[wave], hs0, hs1, lane, HID);
  __syncthreads();
  v16h af = load_afrag(sc[wave], lane);
  v8f mt = wmma16<false>(af, bW3, cb3);                     // cols 0..4 valid
  __syncthreads();

  // ---- cat = [moments(5) | non_invar(8) | 0-pad] as next A tile ----
#pragma unroll
  for (int v = 0; v < 8; ++v) {
    int m = lh * 8 + v;
    float val;
    if (lcol < NM)            val = mt[v];
    else if (lcol < NM + NIN) val = ninv[(size_t)(r0 + m) * NIN + (lcol - NM)];
    else                      val = 0.f;
    sc[wave][m][lcol]      = (half_t)val;
    sc[wave][m][16 + lcol] = (half_t)0;
  }
  __syncthreads();

  // ---- decoder layer 1 ----
  af = load_afrag(sc[wave], lane);
  v8f g0 = relu8(wmma16<false>(af, bD1a, cd1a));
  v8f g1 = relu8(wmma16<true>(af, bD1b, cd1b));
  __syncthreads();
  store_ctile(sc[wave], g0, g1, lane, HID);
  __syncthreads();

  // ---- decoder layer 2 ----
  af = load_afrag(sc[wave], lane);
  g0 = relu8(wmma16<false>(af, bD2a, cd2a));
  g1 = relu8(wmma16<true>(af, bD2b, cd2b));
  __syncthreads();
  store_ctile(sc[wave], g0, g1, lane, HID);
  __syncthreads();

  // ---- output layer ----
  af = load_afrag(sc[wave], lane);
  v8f ot = wmma16<false>(af, bD3, cd3);                     // cols 0..3 valid

  if (lcol < ODIM) {
#pragma unroll
    for (int v = 0; v < 8; ++v) {
      out[(size_t)(r0 + lh * 8 + v) * ODIM + lcol] = ot[v];
    }
  }
}

extern "C" void kernel_launch(void* const* d_in, const int* in_sizes, int n_in,
                              void* d_out, int out_size, void* d_ws, size_t ws_size,
                              hipStream_t stream) {
  const float* invar = (const float*)d_in[0];
  const float* ninv  = (const float*)d_in[1];
  const float* eW1 = (const float*)d_in[2];
  const float* eb1 = (const float*)d_in[3];
  const float* eW2 = (const float*)d_in[4];
  const float* eb2 = (const float*)d_in[5];
  const float* eW3 = (const float*)d_in[6];
  const float* eb3 = (const float*)d_in[7];
  const float* dW1 = (const float*)d_in[8];
  const float* db1 = (const float*)d_in[9];
  const float* dW2 = (const float*)d_in[10];
  const float* db2 = (const float*)d_in[11];
  const float* dW3 = (const float*)d_in[12];
  const float* db3 = (const float*)d_in[13];
  float* out = (float*)d_out;

  const int batch = in_sizes[0] / NB;           // 131072
  const int rows_per_block = WAVES * 16;        // 128
  const int grid = batch / rows_per_block;      // 1024

  perminv_qnn_kernel<<<grid, 256, 0, stream>>>(invar, ninv, eW1, eb1, eW2, eb2, eW3, eb3,
                                               dW1, db1, dW2, db2, dW3, db3, out);
}
